// SMARTMapDecoder_13426067768000
// MI455X (gfx1250) — compile-verified, matrix operand order checked by reading, not run on hardware
//
#include <hip/hip_runtime.h>
#include <hip/hip_bf16.h>

// ---------------------------------------------------------------------------
// Types for CDNA5 WMMA
// ---------------------------------------------------------------------------
typedef __attribute__((ext_vector_type(16))) _Float16 v16h;
typedef __attribute__((ext_vector_type(8)))  _Float16 v8h;
typedef __attribute__((ext_vector_type(4)))  _Float16 v4h;
typedef __attribute__((ext_vector_type(8)))  float    v8f;

union V16U { v16h v; v8h h[2]; };

#define HID_C 128

// ---------------------------------------------------------------------------
// Ordered-key float <-> uint for atomic max on floats
// ---------------------------------------------------------------------------
__device__ __forceinline__ unsigned f2ord(float f) {
  unsigned u = __float_as_uint(f);
  return (u & 0x80000000u) ? ~u : (u | 0x80000000u);
}
__device__ __forceinline__ float ord2f(unsigned k) {
  unsigned u = (k & 0x80000000u) ? (k ^ 0x80000000u) : ~k;
  return __uint_as_float(u);
}

// ---------------------------------------------------------------------------
// Generic WMMA GEMM:  C[M,Ntot] = act( A[M,Kpad(f16)] @ W[K,Ntot(f32)] + bias )
//  block = 256 threads (8 wave32 waves); each wave computes 16 rows x 128 cols.
//  grid = (ceil(M/128), Ntot/128). Kpad multiple of 64; W rows k>=K staged 0.
//  LDS weight tile is stored K-contiguous per column (stride 72 halves) so a
//  B fragment is two aligned ds_load_b128 per 16x16 tile (no scalar gathers).
//  ACT: 0=none 1=relu 2=sigmoid. ACCUM: C += result. OUT16: write f16 to Ch.
// ---------------------------------------------------------------------------
template<int ACT, bool ACCUM, bool OUT16>
__global__ __launch_bounds__(256)
void gemm_wmma(const _Float16* __restrict__ A, int Kpad,
               const float* __restrict__ W, int K, int Ntot,
               const float* __restrict__ bias,
               float* __restrict__ Cf, _Float16* __restrict__ Ch, int M)
{
  __shared__ _Float16 sWc[128 * 72];       // [col][k] col-stride 72 halves, 18 KB
  const int tid  = threadIdx.x;
  const int lane = tid & 31;
  const int wave = tid >> 5;
  const int g    = lane >> 4;      // lane half selects K sub-pattern
  const int nl   = lane & 15;      // row (for A) / col-in-tile (for B/C)
  const int n0   = blockIdx.y * 128;
  const int row0 = blockIdx.x * 128 + wave * 16;

  v8f acc[8];
  for (int t = 0; t < 8; ++t)
    for (int r = 0; r < 8; ++r) acc[t][r] = 0.f;

  int arow = row0 + nl;
  if (arow >= M) arow = M - 1;
  const _Float16* Arow = A + (size_t)arow * Kpad;

  for (int kb = 0; kb < Kpad; kb += 64) {
    // stage weight panel transposed (f32 -> f16, zero-pad K tail)
    for (int idx = tid; idx < 64 * 128; idx += 256) {
      int kk = idx >> 7, nn = idx & 127;
      int gk = kb + kk;
      float wv = (gk < K) ? W[(size_t)gk * Ntot + n0 + nn] : 0.f;
      sWc[nn * 72 + kk] = (_Float16)wv;
    }
    __syncthreads();

    if (kb + 64 < Kpad) __builtin_prefetch(Arow + kb + 64, 0, 0);  // global_prefetch_b8

    for (int k32 = 0; k32 < 64; k32 += 32) {
      // A fragment: 16x32 f16 per ISA layout (two aligned 16B global loads)
      V16U a;
      a.h[0] = *(const v8h*)(Arow + kb + k32 + g * 8);
      a.h[1] = *(const v8h*)(Arow + kb + k32 + 16 + g * 8);
      for (int t = 0; t < 8; ++t) {
        // B fragment: two aligned ds_load_b128 from the K-contiguous tile
        const _Float16* bp = sWc + (t * 16 + nl) * 72 + k32 + g * 8;
        V16U b;
        b.h[0] = *(const v8h*)(bp);
        b.h[1] = *(const v8h*)(bp + 16);
        acc[t] = __builtin_amdgcn_wmma_f32_16x16x32_f16(
            false, a.v, false, b.v, (short)0, acc[t], false, false);
      }
    }
    __syncthreads();
  }

  // epilogue
  for (int t = 0; t < 8; ++t) {
    const int col = n0 + t * 16 + nl;
    const float bv = bias ? bias[col] : 0.f;
    for (int r = 0; r < 8; ++r) {
      const int row = row0 + g * 8 + r;
      if (row >= M) continue;
      float val = acc[t][r] + bv;
      const size_t o = (size_t)row * Ntot + col;
      if (ACCUM) val += Cf[o];
      if (ACT == 1) val = val > 0.f ? val : 0.f;
      if (ACT == 2) val = 1.f / (1.f + __expf(-val));
      if (OUT16) Ch[o] = (_Float16)val; else Cf[o] = val;
    }
  }
}

// ---------------------------------------------------------------------------
// LayerNorm (one wave32 per row), f32 in -> f16 out, optional relu. D<=256.
// ---------------------------------------------------------------------------
__global__ void ln_f32_f16_kernel(const float* __restrict__ in, _Float16* __restrict__ out,
                                  int M, int D, const float* __restrict__ gam,
                                  const float* __restrict__ bet, int relu)
{
  int warp = (blockIdx.x * blockDim.x + threadIdx.x) >> 5;
  int lane = threadIdx.x & 31;
  if (warp >= M) return;
  const float* rp = in + (size_t)warp * D;
  int cnt = D >> 5;
  float v[8];
  float s = 0.f;
  for (int j = 0; j < cnt; ++j) { v[j] = rp[j * 32 + lane]; s += v[j]; }
  for (int o = 16; o; o >>= 1) s += __shfl_xor(s, o, 32);
  float mu = s / (float)D;
  float q = 0.f;
  for (int j = 0; j < cnt; ++j) { float d = v[j] - mu; q += d * d; }
  for (int o = 16; o; o >>= 1) q += __shfl_xor(q, o, 32);
  float rs = rsqrtf(q / (float)D + 1e-5f);
  _Float16* op = out + (size_t)warp * D;
  for (int j = 0; j < cnt; ++j) {
    int c = j * 32 + lane;
    float y = (v[j] - mu) * rs * gam[c] + bet[c];
    if (relu) y = y > 0.f ? y : 0.f;
    op[c] = (_Float16)y;
  }
}

// in-place f16 LayerNorm (fourier hidden)
__global__ void ln_f16_ip_kernel(_Float16* __restrict__ buf, int M, int D,
                                 const float* __restrict__ gam,
                                 const float* __restrict__ bet, int relu)
{
  int warp = (blockIdx.x * blockDim.x + threadIdx.x) >> 5;
  int lane = threadIdx.x & 31;
  if (warp >= M) return;
  _Float16* rp = buf + (size_t)warp * D;
  int cnt = D >> 5;
  float v[8];
  float s = 0.f;
  for (int j = 0; j < cnt; ++j) { v[j] = (float)rp[j * 32 + lane]; s += v[j]; }
  for (int o = 16; o; o >>= 1) s += __shfl_xor(s, o, 32);
  float mu = s / (float)D;
  float q = 0.f;
  for (int j = 0; j < cnt; ++j) { float d = v[j] - mu; q += d * d; }
  for (int o = 16; o; o >>= 1) q += __shfl_xor(q, o, 32);
  float rs = rsqrtf(q / (float)D + 1e-5f);
  for (int j = 0; j < cnt; ++j) {
    int c = j * 32 + lane;
    float y = (v[j] - mu) * rs * gam[c] + bet[c];
    if (relu) y = y > 0.f ? y : 0.f;
    rp[c] = (_Float16)y;
  }
}

// ---------------------------------------------------------------------------
// Elementwise / gather / scatter kernels
// ---------------------------------------------------------------------------
__global__ void fill_f32_kernel(float* p, float v, int n) {
  int i = blockIdx.x * blockDim.x + threadIdx.x;
  if (i < n) p[i] = v;
}
__global__ void fill_u32_kernel(unsigned* p, unsigned v, int n) {
  int i = blockIdx.x * blockDim.x + threadIdx.x;
  if (i < n) p[i] = v;
}
__global__ void copy_f32_kernel(const float* in, float* out, int n) {
  int i = blockIdx.x * blockDim.x + threadIdx.x;
  if (i < n) out[i] = in[i];
}
__global__ void copy_i32_kernel(const int* in, int* out, int n) {
  int i = blockIdx.x * blockDim.x + threadIdx.x;
  if (i < n) out[i] = in[i];
}
__global__ void cvt_pad_kernel(const float* __restrict__ in, _Float16* __restrict__ out,
                               int M, int Kin, int Kpad) {
  int i = blockIdx.x * blockDim.x + threadIdx.x;
  if (i >= M * Kpad) return;
  int m = i / Kpad, k = i - m * Kpad;
  out[i] = (k < Kin) ? (_Float16)in[(size_t)m * Kin + k] : (_Float16)0.f;
}
__global__ void embed_kernel(const float* __restrict__ xtok, const int* __restrict__ token_idx,
                             const float* __restrict__ type_emb, const int* __restrict__ pt_type,
                             const float* __restrict__ pl_type_emb, const int* __restrict__ pl_type,
                             const float* __restrict__ light_emb, const int* __restrict__ pl_light,
                             const int* __restrict__ t2p, float* __restrict__ x, int n) {
  int i = blockIdx.x * blockDim.x + threadIdx.x;
  if (i >= n * HID_C) return;
  int m = i >> 7, c = i & 127;
  x[i] = xtok[(size_t)token_idx[m] * HID_C + c]
       + type_emb[(size_t)pt_type[m] * HID_C + c]
       + pl_type_emb[(size_t)pl_type[m] * HID_C + c]
       + light_emb[(size_t)pl_light[t2p[m]] * HID_C + c];
}
__global__ void edge_geom_kernel(const float* __restrict__ pos, const float* __restrict__ orient,
                                 const int* __restrict__ src, const int* __restrict__ dst,
                                 float* __restrict__ r3, int Ec) {
  int e = blockIdx.x * blockDim.x + threadIdx.x;
  if (e >= Ec) return;
  int s = src[e], d = dst[e];
  float rx = pos[(size_t)s * 2]     - pos[(size_t)d * 2];
  float ry = pos[(size_t)s * 2 + 1] - pos[(size_t)d * 2 + 1];
  const float PI = 3.14159265358979323846f;
  float w = fmodf(orient[s] - orient[d] + PI, 2.f * PI);
  if (w < 0.f) w += 2.f * PI;
  w -= PI;
  float co = cosf(orient[d]), so = sinf(orient[d]);
  float ang = atan2f(co * ry - so * rx, co * rx + so * ry);
  r3[(size_t)e * 3 + 0] = sqrtf(rx * rx + ry * ry);
  r3[(size_t)e * 3 + 1] = ang;
  r3[(size_t)e * 3 + 2] = w;
}
// features for fourier dim di: [cos(r*f*2pi) x64 | sin x64 | r | 0-pad -> 192]
__global__ void edge_feat_kernel(const float* __restrict__ r3, const float* __restrict__ freqs,
                                 int di, _Float16* __restrict__ feat, int Ec) {
  int i = blockIdx.x * blockDim.x + threadIdx.x;
  if (i >= Ec * 48) return;
  int e = i / 48, qq = (i - e * 48) * 4;
  float rv = r3[(size_t)e * 3 + di];
  _Float16* fp = feat + (size_t)e * 192;
  const float TWO_PI = 6.28318530717958647692f;
  v4h o4;
  for (int j = 0; j < 4; ++j) {
    int k = qq + j;
    float val;
    if (k < 64)       val = cosf(rv * freqs[di * 64 + k] * TWO_PI);
    else if (k < 128) val = sinf(rv * freqs[di * 64 + (k - 64)] * TWO_PI);
    else if (k == 128) val = rv;
    else               val = 0.f;
    o4[j] = (_Float16)val;
  }
  *(v4h*)(fp + qq) = o4;
}
__global__ void edge_sim_kernel(const _Float16* __restrict__ qh, const _Float16* __restrict__ kh,
                                const _Float16* __restrict__ krh, const int* __restrict__ src,
                                const int* __restrict__ dst, float* __restrict__ sim,
                                unsigned* __restrict__ segmax, int Ec) {
  int i = blockIdx.x * blockDim.x + threadIdx.x;
  if (i >= Ec * 8) return;
  int e = i >> 3, h = i & 7;
  int s = src[e], d = dst[e];
  const v8h* qp = (const v8h*)(qh  + (size_t)d * HID_C + h * 16);
  const v8h* kp = (const v8h*)(kh  + (size_t)s * HID_C + h * 16);
  const v8h* rp = (const v8h*)(krh + (size_t)e * HID_C + h * 16);
  float acc = 0.f;
  for (int half = 0; half < 2; ++half) {
    v8h qv = qp[half], kv = kp[half], rv = rp[half];
    for (int j = 0; j < 8; ++j)
      acc += (float)qv[j] * ((float)kv[j] + (float)rv[j]);
  }
  acc *= 0.25f;                               // HEAD_DIM^-0.5
  sim[i] = acc;
  atomicMax(&segmax[(size_t)d * 8 + h], f2ord(acc));
}
__global__ void edge_exp_kernel(float* __restrict__ sim, const unsigned* __restrict__ segmax,
                                float* __restrict__ segsum, const int* __restrict__ dst, int Ec) {
  int i = blockIdx.x * blockDim.x + threadIdx.x;
  if (i >= Ec * 8) return;
  int e = i >> 3, h = i & 7;
  int d = dst[e];
  float w = __expf(sim[i] - ord2f(segmax[(size_t)d * 8 + h]));
  sim[i] = w;
  atomicAdd(&segsum[(size_t)d * 8 + h], w);
}
__global__ void edge_agg_kernel(const float* __restrict__ sim, const float* __restrict__ segsum,
                                const _Float16* __restrict__ vh, const _Float16* __restrict__ vrh,
                                const int* __restrict__ src, const int* __restrict__ dst,
                                float* __restrict__ agg, int Ec) {
  int i = blockIdx.x * blockDim.x + threadIdx.x;
  if (i >= Ec * 32) return;
  int e = i >> 5, c4 = (i & 31) * 4;
  int s = src[e], d = dst[e];
  int h = c4 >> 4;
  float a = sim[(size_t)e * 8 + h] / (segsum[(size_t)d * 8 + h] + 1e-16f);
  v4h vv = *(const v4h*)(vh  + (size_t)s * HID_C + c4);
  v4h rv = *(const v4h*)(vrh + (size_t)e * HID_C + c4);
  for (int j = 0; j < 4; ++j) {
    atomicAdd(&agg[(size_t)d * HID_C + c4 + j], a * ((float)vv[j] + (float)rv[j]));
  }
}
__global__ void cat_kernel(const float* __restrict__ agg, const _Float16* __restrict__ xnh,
                           _Float16* __restrict__ cath, int n) {
  int i = blockIdx.x * blockDim.x + threadIdx.x;
  if (i >= n * 256) return;
  int m = i >> 8, c = i & 255;
  cath[i] = (c < 128) ? (_Float16)agg[(size_t)m * HID_C + c]
                      : xnh[(size_t)m * HID_C + (c - 128)];
}
__global__ void gate_kernel(const float* __restrict__ agg, const float* __restrict__ gb,
                            const float* __restrict__ sb, _Float16* __restrict__ uh, int n) {
  int i = blockIdx.x * blockDim.x + threadIdx.x;
  if (i >= n * HID_C) return;
  float a = agg[i];
  uh[i] = (_Float16)(a + gb[i] * (sb[i] - a));
}
// top-10 of 1024 logits per row (one wave32 per row); softmax is monotonic.
__global__ void topk10_kernel(const float* __restrict__ logits, int* __restrict__ out, int rows) {
  int warp = (blockIdx.x * blockDim.x + threadIdx.x) >> 5;
  int lane = threadIdx.x & 31;
  if (warp >= rows) return;
  const float* rp = logits + (size_t)warp * 1024;
  float v[32];
  for (int j = 0; j < 32; ++j) v[j] = rp[j * 32 + lane];
  for (int it = 0; it < 10; ++it) {
    float bv = -3.4e38f; int bi = 0x7fffffff;
    for (int j = 0; j < 32; ++j) {
      int idx = j * 32 + lane;
      if (v[j] > bv || (v[j] == bv && idx < bi)) { bv = v[j]; bi = idx; }
    }
    for (int o = 16; o; o >>= 1) {
      float ov = __shfl_xor(bv, o, 32);
      int   oi = __shfl_xor(bi, o, 32);
      if (ov > bv || (ov == bv && oi < bi)) { bv = ov; bi = oi; }
    }
    if (lane == 0) out[(size_t)warp * 10 + it] = bi;
    if ((bi & 31) == lane) v[bi >> 5] = -3.4e38f;
  }
}

// ---------------------------------------------------------------------------
// Host orchestration
// ---------------------------------------------------------------------------
static inline dim3 g1(long long n) { return dim3((unsigned)((n + 255) / 256)); }

extern "C" void kernel_launch(void* const* d_in, const int* in_sizes, int n_in,
                              void* d_out, int out_size, void* d_ws, size_t ws_size,
                              hipStream_t stream)
{
  (void)n_in; (void)out_size; (void)ws_size;
  const int NP = in_sizes[1];            // 30000
  const int E  = in_sizes[100] / 2;      // 600000
  const int NT = 1024;

  auto F = [&](int i) { return (const float*)d_in[i]; };
  auto I = [&](int i) { return (const int*)d_in[i]; };

  // --- input indices (setup_inputs insertion order, params flattened depth-first)
  const float* pos_pt     = F(0);
  const float* orient_pt  = F(1);
  const float* token_samp = F(2);
  const float* te_W1 = F(3);  const float* te_b1 = F(4);
  const float* te_lg = F(5);  const float* te_lb = F(6);
  const float* te_W2 = F(7);  const float* te_b2 = F(8);
  const float* type_emb = F(9);
  const float* pl_type_emb = F(10);
  const float* light_emb = F(11);
  const float* re_freqs = F(12);
  const float* re_W1 = F(13); const float* re_b1 = F(14);
  const float* re_l1g = F(15); const float* re_l1b = F(16);
  const float* re_W2 = F(17); const float* re_b2 = F(18);
  const float* re_log = F(19); const float* re_lob = F(20);
  const float* re_Wo = F(21); const float* re_bo = F(22);
  const int LBASE = 23;                  // 22 leaves per layer, 3 layers
  const float* hd_W1 = F(89); const float* hd_b1 = F(90);
  const float* hd_lg = F(91); const float* hd_lb = F(92);
  const float* hd_W2 = F(93); const float* hd_b2 = F(94);
  const int* token_idx = I(95);
  const int* pt_type   = I(96);
  const int* pl_type   = I(97);
  const int* pl_light  = I(98);
  const int* t2p       = I(99);
  const int* e_src = I(100);
  const int* e_dst = I(100) + E;

  // --- workspace layout
  char* ws = (char*)d_ws;
  size_t off = 0;
  auto alloc = [&](size_t bytes) { off = (off + 255) & ~(size_t)255; size_t o = off; off += bytes; return o; };
  float*    r3    = (float*)   (ws + alloc((size_t)E * 3 * 4));
  _Float16* feath = (_Float16*)(ws + alloc((size_t)E * 192 * 2));
  _Float16* eh    = (_Float16*)(ws + alloc((size_t)E * HID_C * 2));  // fourier t_h / rn_h
  _Float16* krh   = (_Float16*)(ws + alloc((size_t)E * HID_C * 2));  // also fourier hn_h
  _Float16* vrh   = (_Float16*)(ws + alloc((size_t)E * HID_C * 2));
  float*    rbuf  = (float*)   (ws + alloc((size_t)E * HID_C * 4));  // hsum -> r
  float*    simb  = (float*)   (ws + alloc((size_t)E * 8 * 4));
  unsigned* segmx = (unsigned*)(ws + alloc((size_t)NP * 8 * 4));
  float*    segsm = (float*)   (ws + alloc((size_t)NP * 8 * 4));
  float*    x     = (float*)   (ws + alloc((size_t)NP * HID_C * 4));
  _Float16* xnh   = (_Float16*)(ws + alloc((size_t)NP * HID_C * 2));
  _Float16* qh    = (_Float16*)(ws + alloc((size_t)NP * HID_C * 2));
  _Float16* kh    = (_Float16*)(ws + alloc((size_t)NP * HID_C * 2));
  _Float16* vh    = (_Float16*)(ws + alloc((size_t)NP * HID_C * 2));
  float*    agg   = (float*)   (ws + alloc((size_t)NP * HID_C * 4));
  float*    sb    = (float*)   (ws + alloc((size_t)NP * HID_C * 4));
  float*    gb    = (float*)   (ws + alloc((size_t)NP * HID_C * 4));
  _Float16* cath  = (_Float16*)(ws + alloc((size_t)NP * 256 * 2));
  _Float16* uh    = (_Float16*)(ws + alloc((size_t)NP * HID_C * 2));
  _Float16* ffh   = (_Float16*)(ws + alloc((size_t)NP * 512 * 2));
  _Float16* tsh   = (_Float16*)(ws + alloc((size_t)NT * 64 * 2));
  float*    tok1  = (float*)   (ws + alloc((size_t)NT * 256 * 4));
  _Float16* tok1h = (_Float16*)(ws + alloc((size_t)NT * 256 * 2));
  float*    xtok  = (float*)   (ws + alloc((size_t)NT * HID_C * 4));

  // --- output layout (concatenated, float32-sized elements)
  float* outf = (float*)d_out;
  int*   outi = (int*)d_out;
  const size_t X_OUT = 0;
  const size_t IDX_OUT = (size_t)NP * HID_C;             // 3,840,000
  const size_t LOG_OUT = IDX_OUT + (size_t)NP * 10;      // 4,140,000
  const size_t TOK_OUT = LOG_OUT + (size_t)NP * NT;      // 34,860,000

  const dim3 B(256);

  // ===== 1) token embedding MLP =====
  cvt_pad_kernel<<<g1((long long)NT * 64), B, 0, stream>>>(token_samp, tsh, NT, 22, 64);
  gemm_wmma<0,false,false><<<dim3((NT+127)/128, 2), B, 0, stream>>>(tsh, 64, te_W1, 22, 256, te_b1, tok1, nullptr, NT);
  ln_f32_f16_kernel<<<g1((long long)NT * 32), B, 0, stream>>>(tok1, tok1h, NT, 256, te_lg, te_lb, 1);
  gemm_wmma<0,false,false><<<dim3((NT+127)/128, 1), B, 0, stream>>>(tok1h, 256, te_W2, 256, 128, te_b2, xtok, nullptr, NT);
  embed_kernel<<<g1((long long)NP * 128), B, 0, stream>>>(xtok, token_idx, type_emb, pt_type,
      pl_type_emb, pl_type, light_emb, pl_light, t2p, x, NP);

  // ===== 2) edge geometry + fourier rel-pos embedding =====
  edge_geom_kernel<<<g1(E), B, 0, stream>>>(pos_pt, orient_pt, e_src, e_dst, r3, E);
  const dim3 GE((E + 127) / 128, 1);
  for (int di = 0; di < 3; ++di) {
    edge_feat_kernel<<<g1((long long)E * 48), B, 0, stream>>>(r3, re_freqs, di, feath, E);
    gemm_wmma<0,false,true><<<GE, B, 0, stream>>>(feath, 192, re_W1 + (size_t)di*129*128, 129, 128,
                                                  re_b1 + di*128, nullptr, eh, E);
    ln_f16_ip_kernel<<<g1((long long)E * 32), B, 0, stream>>>(eh, E, 128, re_l1g + di*128, re_l1b + di*128, 1);
    if (di == 0)
      gemm_wmma<0,false,false><<<GE, B, 0, stream>>>(eh, 128, re_W2 + (size_t)di*128*128, 128, 128,
                                                     re_b2 + di*128, rbuf, nullptr, E);
    else
      gemm_wmma<0,true,false><<<GE, B, 0, stream>>>(eh, 128, re_W2 + (size_t)di*128*128, 128, 128,
                                                    re_b2 + di*128, rbuf, nullptr, E);
  }
  ln_f32_f16_kernel<<<g1((long long)E * 32), B, 0, stream>>>(rbuf, krh, E, 128, re_log, re_lob, 1);
  gemm_wmma<0,false,false><<<GE, B, 0, stream>>>(krh, 128, re_Wo, 128, 128, re_bo, rbuf, nullptr, E);

  // ===== 3) attention layers =====
  const dim3 GN((NP + 127) / 128, 1);
  for (int l = 0; l < 3; ++l) {
    const int P = LBASE + 22 * l;
    const float *ln_xg=F(P+0), *ln_xb=F(P+1), *ln_rg=F(P+2), *ln_rb=F(P+3);
    const float *Wq=F(P+4), *bq=F(P+5), *Wk=F(P+6), *Wv=F(P+7);
    const float *Wkr=F(P+8), *Wvr=F(P+9), *Ws=F(P+10), *bs=F(P+11);
    const float *Wg=F(P+12), *bg=F(P+13), *Wo=F(P+14), *bo=F(P+15);
    const float *ln_fg=F(P+16), *ln_fb=F(P+17);
    const float *W1=F(P+18), *b1=F(P+19), *W2=F(P+20), *b2=F(P+21);

    ln_f32_f16_kernel<<<g1((long long)NP * 32), B, 0, stream>>>(x, xnh, NP, 128, ln_xg, ln_xb, 0);
    ln_f32_f16_kernel<<<g1((long long)E * 32), B, 0, stream>>>(rbuf, eh, E, 128, ln_rg, ln_rb, 0);

    gemm_wmma<0,false,true><<<GN, B, 0, stream>>>(xnh, 128, Wq, 128, 128, bq, nullptr, qh, NP);
    gemm_wmma<0,false,true><<<GN, B, 0, stream>>>(xnh, 128, Wk, 128, 128, nullptr, nullptr, kh, NP);
    gemm_wmma<0,false,true><<<GN, B, 0, stream>>>(xnh, 128, Wv, 128, 128, nullptr, nullptr, vh, NP);
    gemm_wmma<0,false,true><<<GE, B, 0, stream>>>(eh, 128, Wkr, 128, 128, nullptr, nullptr, krh, E);
    gemm_wmma<0,false,true><<<GE, B, 0, stream>>>(eh, 128, Wvr, 128, 128, nullptr, nullptr, vrh, E);

    fill_u32_kernel<<<g1(NP * 8), B, 0, stream>>>(segmx, 0u, NP * 8);
    fill_f32_kernel<<<g1(NP * 8), B, 0, stream>>>(segsm, 0.f, NP * 8);
    fill_f32_kernel<<<g1((long long)NP * 128), B, 0, stream>>>(agg, 0.f, NP * 128);
    edge_sim_kernel<<<g1((long long)E * 8), B, 0, stream>>>(qh, kh, krh, e_src, e_dst, simb, segmx, E);
    edge_exp_kernel<<<g1((long long)E * 8), B, 0, stream>>>(simb, segmx, segsm, e_dst, E);
    edge_agg_kernel<<<g1((long long)E * 32), B, 0, stream>>>(simb, segsm, vh, vrh, e_src, e_dst, agg, E);

    cat_kernel<<<g1((long long)NP * 256), B, 0, stream>>>(agg, xnh, cath, NP);
    gemm_wmma<2,false,false><<<GN, B, 0, stream>>>(cath, 256, Wg, 256, 128, bg, gb, nullptr, NP);
    gemm_wmma<0,false,false><<<GN, B, 0, stream>>>(xnh, 128, Ws, 128, 128, bs, sb, nullptr, NP);
    gate_kernel<<<g1((long long)NP * 128), B, 0, stream>>>(agg, gb, sb, uh, NP);
    gemm_wmma<0,true,false><<<GN, B, 0, stream>>>(uh, 128, Wo, 128, 128, bo, x, nullptr, NP);

    ln_f32_f16_kernel<<<g1((long long)NP * 32), B, 0, stream>>>(x, xnh, NP, 128, ln_fg, ln_fb, 0);
    gemm_wmma<1,false,true><<<dim3((NP+127)/128, 4), B, 0, stream>>>(xnh, 128, W1, 128, 512, b1, nullptr, ffh, NP);
    gemm_wmma<0,true,false><<<GN, B, 0, stream>>>(ffh, 512, W2, 512, 128, b2, x, nullptr, NP);
  }

  // ===== 4) outputs: x_pt, head logits, top-10, token_idx =====
  copy_f32_kernel<<<g1((long long)NP * 128), B, 0, stream>>>(x, outf + X_OUT, NP * 128);
  cvt_pad_kernel<<<g1((long long)NP * 128), B, 0, stream>>>(x, xnh, NP, 128, 128);
  gemm_wmma<0,false,false><<<GN, B, 0, stream>>>(xnh, 128, hd_W1, 128, 128, hd_b1, sb, nullptr, NP);
  ln_f32_f16_kernel<<<g1((long long)NP * 32), B, 0, stream>>>(sb, kh, NP, 128, hd_lg, hd_lb, 1);
  gemm_wmma<0,false,false><<<dim3((NP+127)/128, 8), B, 0, stream>>>(kh, 128, hd_W2, 128, 1024, hd_b2,
                                                                    outf + LOG_OUT, nullptr, NP);
  topk10_kernel<<<g1((long long)NP * 32), B, 0, stream>>>(outf + LOG_OUT, outi + IDX_OUT, NP);
  copy_i32_kernel<<<g1(NP), B, 0, stream>>>(token_idx, outi + TOK_OUT, NP);
}
